// TempCtxAttention_70042326663622
// MI455X (gfx1250) — matrix operand order repaired and checked by reading, not compiled, not run on hardware
//
#include <hip/hip_runtime.h>
#include <hip/hip_bf16.h>

// ---------------------------------------------------------------------------
// Problem constants (from reference): N=256, M=256, KLEN=128, D_MODEL=256,
// H=8 heads, DK=32.  All matmuls run on v_wmma_f32_16x16x32_bf16; bf16 tile
// staging into LDS uses gfx1250 async global->LDS copies (ASYNCcnt).
// ---------------------------------------------------------------------------

typedef __bf16 bf16_t;
typedef __attribute__((ext_vector_type(16))) __bf16   v16bf;
typedef __attribute__((ext_vector_type(8)))  float    v8f;
typedef __attribute__((ext_vector_type(8)))  unsigned v8u;

#define WMMA_BF16(a, b, c) \
    __builtin_amdgcn_wmma_f32_16x16x32_bf16(false, (a), false, (b), (short)0, (c), false, false)

// Async copy of 16 bytes global -> LDS (per lane).  LDS aperture maps
// addr[31:0] directly to the LDS offset, so truncating the generic pointer
// gives the ds address.  Tracked with ASYNCcnt.
__device__ __forceinline__ void async_b128_to_lds(const void* gptr, void* lptr) {
    unsigned laddr = (unsigned)(unsigned long long)lptr;
    asm volatile("global_load_async_to_lds_b128 %0, %1, off"
                 :: "v"(laddr), "v"((unsigned long long)gptr) : "memory");
}
__device__ __forceinline__ void wait_async0() {
    asm volatile("s_wait_asynccnt 0" ::: "memory");
}

// A-matrix fragment (16x32 bf16, MxK).  Row M = lane&15, K pairs per the ISA
// table: VGPR j holds K = 16*(j>>2) + 8*(lane>>4) + 2*(j&3) (+0,+1).
__device__ __forceinline__ v16bf load_A_frag(const bf16_t* base, int stride, int lane) {
    const int r  = lane & 15;
    const int hh = lane >> 4;
    const bf16_t* rp = base + r * stride;
    v8u u;
#pragma unroll
    for (int j = 0; j < 8; ++j) {
        const int kk = 16 * (j >> 2) + 8 * hh + 2 * (j & 3);
        u[j] = *(const unsigned*)(rp + kk);
    }
    return __builtin_bit_cast(v16bf, u);
}

// B-matrix fragment (32x16 bf16, KxN) read from a row-major *transposed* B
// (i.e. Bt[n][k]): column N = lane&15, VGPR j holds K = 16*(j>>2) + 8*(lane>>4) + 2*(j&3).
__device__ __forceinline__ v16bf load_B_frag(const bf16_t* baseT, int stride, int lane) {
    const int c  = lane & 15;
    const int hh = lane >> 4;
    const bf16_t* cp = baseT + c * stride;
    v8u u;
#pragma unroll
    for (int j = 0; j < 8; ++j) {
        const int kk = 16 * (j >> 2) + 8 * hh + 2 * (j & 3);
        u[j] = *(const unsigned*)(cp + kk);
    }
    return __builtin_bit_cast(v16bf, u);
}

// ---------------------------------------------------------------------------
// f32 -> bf16 conversion (weights)
// ---------------------------------------------------------------------------
__global__ __launch_bounds__(256) void cvt_f32_bf16(const float* __restrict__ in,
                                                    bf16_t* __restrict__ out, int n) {
    int i = blockIdx.x * blockDim.x + threadIdx.x;
    if (i < n) out[i] = (bf16_t)in[i];
}

// ---------------------------------------------------------------------------
// GEMM: Out[r][c] = ((sum_d A[r][d] * W[c][d]) + bias[c]) * scale
//   A: [R x 256]   (f32 if IN_F32, else bf16), row-major
//   W: [256 x 256] bf16 row-major ( == Bt layout for WMMA )
//   Out: bf16 or f32; TRANS_V stores vproj as [m][256][128] for the P@V GEMM.
// Block: 256 threads (8 waves). Block tile 64 rows x 64 cols; each wave owns a
// 16x32 tile (2 WMMA accumulators), K loop = 8 chunks of 32 staged in LDS.
// LDS row stride 40 bf16 = 80 B: 16B-aligned rows (async b128 legal) and
// conflict-free fragment gathers (banks 20*r mod 64 all distinct, r=0..15).
// ---------------------------------------------------------------------------
#define GSTR 40

template <bool IN_F32, bool OUT_F32, bool TRANS_V>
__global__ __launch_bounds__(256) void gemm256_bias(const void* __restrict__ Ain,
                                                    const bf16_t* __restrict__ Wt,
                                                    const float* __restrict__ bias,
                                                    void* __restrict__ Out,
                                                    int R, float scale) {
    __shared__ __align__(16) bf16_t As[64 * GSTR];
    __shared__ __align__(16) bf16_t Bs[64 * GSTR];

    const int t       = threadIdx.x;
    const int lane    = t & 31;
    const int wave    = t >> 5;
    const int rowBase = blockIdx.x * 64;
    const int colBase = blockIdx.y * 64;
    const int rt      = wave >> 1;          // row tile 0..3
    const int ctBase  = (wave & 1) * 2;     // col tiles {0,1} or {2,3}

    const int lrow = t >> 2;                // staging: row/col 0..63
    const int lks  = (t & 3) * 8;           // 8 bf16 (16 B) each

    v8f acc0 = {};
    v8f acc1 = {};

#pragma unroll
    for (int kc = 0; kc < 256; kc += 32) {
        // stage A chunk
        if (IN_F32) {
            // convert f32 -> bf16 on the fly (VALU path)
            const float* Af = (const float*)Ain + (size_t)(rowBase + lrow) * 256 + kc + lks;
#pragma unroll
            for (int e = 0; e < 8; ++e) As[lrow * GSTR + lks + e] = (bf16_t)Af[e];
        } else {
            // bf16 source: async global -> LDS copy, no VGPR roundtrip
            const bf16_t* Ab = (const bf16_t*)Ain + (size_t)(rowBase + lrow) * 256 + kc + lks;
            async_b128_to_lds(Ab, &As[lrow * GSTR + lks]);
        }
        // stage W chunk (always bf16, already Bt layout [col][k]): async copy
        const bf16_t* Wb = Wt + (size_t)(colBase + lrow) * 256 + kc + lks;
        async_b128_to_lds(Wb, &Bs[lrow * GSTR + lks]);
        wait_async0();
        __syncthreads();

        v16bf af = load_A_frag(As + rt * 16 * GSTR, GSTR, lane);
        v16bf b0 = load_B_frag(Bs + (ctBase + 0) * 16 * GSTR, GSTR, lane);
        v16bf b1 = load_B_frag(Bs + (ctBase + 1) * 16 * GSTR, GSTR, lane);
        acc0 = WMMA_BF16(af, b0, acc0);
        acc1 = WMMA_BF16(af, b1, acc1);
        __syncthreads();
    }

    // epilogue: C/D layout row M = r + 8*(lane>>4), col N = lane&15
    const int hh = lane >> 4;
    const int c0 = colBase + (ctBase + 0) * 16 + (lane & 15);
    const int c1 = colBase + (ctBase + 1) * 16 + (lane & 15);
    const float bia0 = bias[c0];
    const float bia1 = bias[c1];
#pragma unroll
    for (int r = 0; r < 8; ++r) {
        const int grow = rowBase + rt * 16 + 8 * hh + r;
        const float v0 = (acc0[r] + bia0) * scale;
        const float v1 = (acc1[r] + bia1) * scale;
        if (TRANS_V) {
            // grow = m*128 + klen ; store vT[m][col][klen]
            const int m  = grow >> 7;
            const int kl = grow & 127;
            bf16_t* ob = (bf16_t*)Out;
            ob[((size_t)m * 256 + c0) * 128 + kl] = (bf16_t)v0;
            ob[((size_t)m * 256 + c1) * 128 + kl] = (bf16_t)v1;
        } else if (OUT_F32) {
            float* of = (float*)Out;
            of[(size_t)grow * 256 + c0] = v0;
            of[(size_t)grow * 256 + c1] = v1;
        } else {
            bf16_t* ob = (bf16_t*)Out;
            ob[(size_t)grow * 256 + c0] = (bf16_t)v0;
            ob[(size_t)grow * 256 + c1] = (bf16_t)v1;
        }
    }
}

// ---------------------------------------------------------------------------
// Attention core: one workgroup per (16-query tile, m); wave w == head w.
//   qb  : [256][256] bf16, pre-scaled by 1/sqrt(dk)
//   kp  : [m][128][256] bf16  (row-major kproj; directly Bt for Q@K^T)
//   vT  : [m][256][128] bf16  (transposed vproj; directly Bt for P@V)
//   ctx : [n*256+m][256] bf16
// Per wave: 8 WMMA for scores (16x128), LDS softmax, 8 WMMA for P@V.
// Q tile staged with async b128 copies (stride 264 bf16 = 528 B: 16B-aligned,
// banks 4r mod 64 distinct for the 16 gathered rows).
// ---------------------------------------------------------------------------
#define QSTR 264

__global__ __launch_bounds__(256) void attn_core(const bf16_t* __restrict__ qb,
                                                 const bf16_t* __restrict__ kp,
                                                 const bf16_t* __restrict__ vT,
                                                 bf16_t* __restrict__ ctx) {
    __shared__ __align__(16) bf16_t Qs[16 * QSTR];      // 16 rows x 256 (+pad)
    __shared__ __align__(16) float  Ss[8][16 * 130];    // per-head scores 16x128 (+pad)
    __shared__ __align__(16) bf16_t Ps[8][16 * 130];    // per-head probabilities bf16

    const int t     = threadIdx.x;
    const int lane  = t & 31;
    const int h     = t >> 5;        // head
    const int nBase = blockIdx.x * 16;
    const int m     = blockIdx.y;
    const int hh    = lane >> 4;
    const int ln    = lane & 15;

    // stage Q tile (all heads) into LDS: 2 async b128 per thread
    {
        const int row = t >> 4;
        const int ks  = (t & 15) * 16;
        const bf16_t* src = qb + (size_t)(nBase + row) * 256 + ks;
        async_b128_to_lds(src,     &Qs[row * QSTR + ks]);
        async_b128_to_lds(src + 8, &Qs[row * QSTR + ks + 8]);
    }
    wait_async0();
    __syncthreads();

    // ---- scores: S[16 x 128] = Qh (16x32) @ Kh^T, one WMMA per 16-col tile
    const v16bf aq = load_A_frag(Qs + h * 32, QSTR, lane);
    const bf16_t* kbase = kp + (size_t)m * 32768 + h * 32;   // [klen][256] slice
    v8f sc[8];
#pragma unroll
    for (int tt = 0; tt < 8; ++tt) {
        v16bf bk = load_B_frag(kbase + (size_t)(tt * 16) * 256, 256, lane);
        v8f z = {};
        sc[tt] = WMMA_BF16(aq, bk, z);
    }
    // dump scores to LDS
    float* S = Ss[h];
#pragma unroll
    for (int tt = 0; tt < 8; ++tt)
#pragma unroll
        for (int r = 0; r < 8; ++r)
            S[(8 * hh + r) * 130 + tt * 16 + ln] = sc[tt][r];
    __syncthreads();

    // ---- softmax over the 128 key positions: 2 lanes per row (64 each)
    {
        const int row = ln;
        float*  rp = S + row * 130 + hh * 64;
        bf16_t* pp = Ps[h] + row * 130 + hh * 64;
        float mx = -1e30f;
#pragma unroll 4
        for (int i = 0; i < 64; ++i) mx = fmaxf(mx, rp[i]);
        mx = fmaxf(mx, __shfl_xor(mx, 16, 32));
        float sum = 0.f;
#pragma unroll 4
        for (int i = 0; i < 64; ++i) {
            float e = __expf(rp[i] - mx);
            sum += e;
            pp[i] = (bf16_t)e;
        }
        sum += __shfl_xor(sum, 16, 32);
        const float inv = 1.f / sum;
#pragma unroll 4
        for (int i = 0; i < 64; ++i) pp[i] = (bf16_t)((float)pp[i] * inv);
    }
    __syncthreads();

    // ---- ctx: C[16 x 32] = P (16x128) @ Vh (128x32), K chunks of 32
    v8f ca0 = {};
    v8f ca1 = {};
    const bf16_t* vbase = vT + (size_t)m * 32768 + (size_t)(h * 32) * 128;  // [32][128]
#pragma unroll
    for (int kc = 0; kc < 128; kc += 32) {
        v16bf ap = load_A_frag(Ps[h] + kc, 130, lane);
        v16bf b0 = load_B_frag(vbase + kc, 128, lane);
        v16bf b1 = load_B_frag(vbase + (size_t)16 * 128 + kc, 128, lane);
        ca0 = WMMA_BF16(ap, b0, ca0);
        ca1 = WMMA_BF16(ap, b1, ca1);
    }
    // store ctx[(n*256+m)][h*32 + d] as bf16 for the output projection
#pragma unroll
    for (int r = 0; r < 8; ++r) {
        const int nl = 8 * hh + r;
        const size_t rowOff = ((size_t)(nBase + nl) * 256 + m) * 256;
        ctx[rowOff + h * 32 + 0 * 16 + ln] = (bf16_t)ca0[r];
        ctx[rowOff + h * 32 + 1 * 16 + ln] = (bf16_t)ca1[r];
    }
}

// ---------------------------------------------------------------------------
// Host-side orchestration
// ---------------------------------------------------------------------------
extern "C" void kernel_launch(void* const* d_in, const int* in_sizes, int n_in,
                              void* d_out, int out_size, void* d_ws, size_t ws_size,
                              hipStream_t stream) {
    (void)in_sizes; (void)n_in; (void)out_size;

    const float* query = (const float*)d_in[0];
    const float* key   = (const float*)d_in[1];
    const float* value = (const float*)d_in[2];
    const float* Wq = (const float*)d_in[3];  const float* bq = (const float*)d_in[4];
    const float* Wk = (const float*)d_in[5];  const float* bk = (const float*)d_in[6];
    const float* Wv = (const float*)d_in[7];  const float* bv = (const float*)d_in[8];
    const float* Wo = (const float*)d_in[9];  const float* bo = (const float*)d_in[10];
    float* out = (float*)d_out;

    // workspace layout (bytes)
    const size_t W_BYTES   = 256 * 256 * sizeof(bf16_t);           // 128 KB each
    const size_t Q_BYTES   = 256 * 256 * sizeof(bf16_t);           // 128 KB
    const size_t KV_BYTES  = (size_t)32768 * 256 * sizeof(bf16_t); // 16 MB each
    const size_t CTX_BYTES = (size_t)65536 * 256 * sizeof(bf16_t); // 32 MB
    const size_t NEED = 4 * W_BYTES + Q_BYTES + 2 * KV_BYTES + CTX_BYTES;
    if (ws_size < NEED) return;

    char* ws = (char*)d_ws;
    size_t off = 0;
    bf16_t* Wq_b = (bf16_t*)(ws + off); off += W_BYTES;
    bf16_t* Wk_b = (bf16_t*)(ws + off); off += W_BYTES;
    bf16_t* Wv_b = (bf16_t*)(ws + off); off += W_BYTES;
    bf16_t* Wo_b = (bf16_t*)(ws + off); off += W_BYTES;
    bf16_t* qb   = (bf16_t*)(ws + off); off += Q_BYTES;
    bf16_t* kp   = (bf16_t*)(ws + off); off += KV_BYTES;
    bf16_t* vT   = (bf16_t*)(ws + off); off += KV_BYTES;
    bf16_t* ctx  = (bf16_t*)(ws + off); off += CTX_BYTES;

    // 1) weights -> bf16
    cvt_f32_bf16<<<256, 256, 0, stream>>>(Wq, Wq_b, 65536);
    cvt_f32_bf16<<<256, 256, 0, stream>>>(Wk, Wk_b, 65536);
    cvt_f32_bf16<<<256, 256, 0, stream>>>(Wv, Wv_b, 65536);
    cvt_f32_bf16<<<256, 256, 0, stream>>>(Wo, Wo_b, 65536);

    // 2) q = (query @ Wq^T + bq) * 1/sqrt(32)   [fold attention scale into q]
    gemm256_bias<true, false, false><<<dim3(4, 4), 256, 0, stream>>>(
        query, Wq_b, bq, qb, 256, 0.17677669529663687f);

    // 3) kproj = key @ Wk^T + bk  -> [m][128][256] bf16
    gemm256_bias<true, false, false><<<dim3(512, 4), 256, 0, stream>>>(
        key, Wk_b, bk, kp, 32768, 1.0f);

    // 4) vproj = value @ Wv^T + bv  -> transposed store [m][256][128] bf16
    gemm256_bias<true, false, true><<<dim3(512, 4), 256, 0, stream>>>(
        value, Wv_b, bv, vT, 32768, 1.0f);

    // 5) attention core: grid (n-tiles, m)
    attn_core<<<dim3(16, 256), 256, 0, stream>>>(qb, kp, vT, ctx);

    // 6) out = ctx @ Wo^T + bo  (f32 output)
    gemm256_bias<false, true, false><<<dim3(1024, 4), 256, 0, stream>>>(
        ctx, Wo_b, bo, out, 65536, 1.0f);
}
